// ReversedAttention_15333033247143
// MI455X (gfx1250) — matrix operand order repaired and checked by reading, not compiled, not run on hardware
//
#include <hip/hip_runtime.h>
#include <hip/hip_bf16.h>
#include <math.h>

// ---------------------------------------------------------------------------
// ReversedAttention for MI455X (gfx1250), wave32 + WMMA f16 (f32 accumulate).
//   1) qkv    = x @ W_qkv  -> stored as f16     (WMMA GEMM, f16 out)
//   2) hs     = x @ W_hs    (N=12, masked)      (WMMA GEMM)
//   3) mc1    = x @ W_mc1                       (WMMA GEMM)
//   4) gate: p=softmax(hs*10), mc=sigmoid(gelu(mc1)@W_mc2), m_out=m*mc,
//            g[b,h,n]=p*m_out                   (scalar)
//   5) l[b,h,i] = sum_j exp(scale*q_i.k_j)      (WMMA + shfl reduce)
//   6) out[b,h,j,:] = sum_i exp(s_ij)*g_i/l_i * v[i,:]  (WMMA, LDS re-layout)
//      + per-head prune partials (deterministic, no atomics)
//   7) prune_mask[b,j] = sum_h partials         (fixed-order reduce)
//   8) y = out_reshaped @ W_proj + b_proj       (WMMA GEMM)
// ---------------------------------------------------------------------------

typedef __attribute__((ext_vector_type(16))) _Float16 v16h;
typedef __attribute__((ext_vector_type(8)))  float    v8f;

union HU { v16h h; unsigned int u[8]; };

#define B_   16
#define N_   576
#define C_   768
#define H_   12
#define DH_  64
#define C3_  2304
#define BN_  (B_ * N_)
#define SCALE_ 0.125f  // 64^-0.5

// Load a 16x32 f16 A-fragment row slice from a row-major f16 row pointer.
// Lane row = l15; abase = (lane>>4)*8. Pairs are contiguous -> u32 loads.
__device__ __forceinline__ v16h load_afrag_h(const _Float16* row, int abase) {
  HU t;
#pragma unroll
  for (int v = 0; v < 8; ++v) {
    const int kk = 2 * abase + 2 * v;
    t.u[v] = *(const unsigned int*)(row + kk);
  }
  return t.h;
}

// Load a 32x16 f16 B-fragment column from a row-major f16 row pointer where
// the K dimension is contiguous in memory (i.e. B[k][n] == row[k]).
// khalf = (lane>>4)*16 -> 16 contiguous halfs.
__device__ __forceinline__ v16h load_bfrag_contig_h(const _Float16* row, int khalf) {
  HU t;
#pragma unroll
  for (int v = 0; v < 8; ++v)
    t.u[v] = *(const unsigned int*)(row + khalf + 2 * v);
  return t.h;
}

// ---------------------------------------------------------------------------
// Generic WMMA GEMM: C[M,N] = A[M,K] @ B[K,N] (+bias). A,B row-major f32.
// One wave per 16x16 tile. K % 32 == 0, M % 16 == 0. GUARD: mask columns >= N
// (branch-free: clamped index + 0/1 multiply). HOUT: write f16 instead of f32.
// ---------------------------------------------------------------------------
template <bool GUARD, bool HOUT>
__global__ __launch_bounds__(256)
void gemm16_wmma(const float* __restrict__ A, const float* __restrict__ Bm,
                 float* __restrict__ Cf, _Float16* __restrict__ Ch,
                 const float* __restrict__ bias,
                 int M, int N, int K, int tiles_n) {
  const int wave = (int)((blockIdx.x * blockDim.x + threadIdx.x) >> 5);
  const int lane = (int)(threadIdx.x & 31);
  const int tm = wave / tiles_n;
  const int tn = wave % tiles_n;
  const int m0 = tm * 16, n0 = tn * 16;
  if (m0 >= M) return;

  const int l15   = lane & 15;
  const int half  = lane >> 4;          // 0 or 1
  const int abase = half * 8;           // A-fragment K base
  const int khalf = half * 16;          // B-fragment K base
  const int arow  = m0 + l15;
  int coln        = n0 + l15;
  float msk = 1.0f;
  if (GUARD) {
    if (coln >= N) { coln = N - 1; msk = 0.0f; }
  }

  const float* ap = A + (size_t)arow * K;
  const float* bp = Bm + (size_t)khalf * N + coln;   // advances by 32*N per chunk

  v8f acc = {};
  for (int k0 = 0; k0 < K; k0 += 32) {
    v16h af, bf;
#pragma unroll
    for (int v = 0; v < 8; ++v) {
      // A 16x32 f16 layout: lanes 0-15: K = 0..15 ; lanes 16-31: 16..31
      const int kk = 2 * abase + 2 * v;
      af[2 * v]     = (_Float16)ap[kk];
      af[2 * v + 1] = (_Float16)ap[kk + 1];
    }
#pragma unroll
    for (int v = 0; v < 8; ++v) {
      // B 32x16 f16 layout: lanes 0-15 hold K=0..15, lanes 16-31 hold K=16..31
      float b0 = bp[(size_t)(2 * v) * N];
      float b1 = bp[(size_t)(2 * v + 1) * N];
      if (GUARD) { b0 *= msk; b1 *= msk; }
      bf[2 * v]     = (_Float16)b0;
      bf[2 * v + 1] = (_Float16)b1;
    }
    acc = __builtin_amdgcn_wmma_f32_16x16x32_f16(false, af, false, bf,
                                                 (short)0, acc, false, false);
    ap += 32;
    bp += (size_t)32 * N;
  }

  if (!GUARD || msk != 0.0f) {
    const float bv = bias ? bias[coln] : 0.0f;
    const int rbase = m0 + half * 8;    // C layout: VGPR r -> M = r (+8 hi half)
#pragma unroll
    for (int r = 0; r < 8; ++r) {
      if (HOUT)
        Ch[(size_t)(rbase + r) * N + coln] = (_Float16)(acc[r]);
      else
        Cf[(size_t)(rbase + r) * N + coln] = acc[r] + bv;
    }
  }
}

// ---------------------------------------------------------------------------
// Gate kernel: one thread per token row (B*N rows).
// ---------------------------------------------------------------------------
__global__ __launch_bounds__(256)
void gate_kernel(const float* __restrict__ hs, const float* __restrict__ mc1,
                 const float* __restrict__ W_mc2, const float* __restrict__ m_in,
                 float* __restrict__ m_out, float* __restrict__ g) {
  const int r = (int)(blockIdx.x * blockDim.x + threadIdx.x);
  if (r >= BN_) return;

  // head-selector softmax, temp = 0.1 (logits * 10)
  float lo[H_], mx = -1e30f;
#pragma unroll
  for (int h = 0; h < H_; ++h) {
    const float t = hs[(size_t)r * H_ + h] * 10.0f;
    lo[h] = t;
    mx = fmaxf(mx, t);
  }
  float s = 0.0f;
#pragma unroll
  for (int h = 0; h < H_; ++h) { lo[h] = __expf(lo[h] - mx); s += lo[h]; }
  const float inv = 1.0f / s;

  // message controller: gelu(mc1) @ W_mc2, sigmoid
  float acc = 0.0f;
  for (int k = 0; k < C_ / 2; ++k) {
    const float xg = mc1[(size_t)r * (C_ / 2) + k];
    const float ge = 0.5f * xg * (1.0f + erff(xg * 0.70710678118f));
    acc += ge * W_mc2[k];
  }
  const float mc = 1.0f / (1.0f + __expf(-acc));
  const float mo = m_in[r] * mc;
  m_out[r] = mo;

  const int b = r / N_, n = r % N_;
#pragma unroll
  for (int h = 0; h < H_; ++h)
    g[((size_t)(b * H_ + h)) * N_ + n] = lo[h] * inv * mo;
}

// ---------------------------------------------------------------------------
// Row-sum stats: l[b,h,i] = sum_j exp(scale * q_i . k_j).
// One wave per (b,h, 16-row i-tile). qkv is f16 [B*N, 3C] row-major.
// Scores ~N(0, 2.4^2) so exp without max subtraction is safe in f32.
// ---------------------------------------------------------------------------
__global__ __launch_bounds__(256)
void attn_stats(const _Float16* __restrict__ qkv, float* __restrict__ lsum) {
  const int wave = (int)((blockIdx.x * blockDim.x + threadIdx.x) >> 5);
  const int lane = (int)(threadIdx.x & 31);
  const int tiles_i = N_ / 16;                 // 36
  const int bh = wave / tiles_i;
  if (bh >= B_ * H_) return;
  const int itile = wave % tiles_i;
  const int b = bh / H_, h = bh % H_;
  const int i0 = itile * 16;

  const int l15 = lane & 15, half = lane >> 4;
  const int abase = half * 8, khalf = half * 16;

  // A-fragments: q rows i0..i0+15, K = d (64) -> two 16x32 fragments
  const _Float16* qr = qkv + (size_t)(b * N_ + i0 + l15) * C3_ + h * DH_;
  const v16h aq0 = load_afrag_h(qr, abase);
  const v16h aq1 = load_afrag_h(qr + 32, abase);

  float ps[8];
#pragma unroll
  for (int r = 0; r < 8; ++r) ps[r] = 0.0f;

  for (int j0 = 0; j0 < N_; j0 += 16) {
    // B-fragments: B[d][j] = k[j][d]; K=d contiguous along the k row
    const _Float16* kr = qkv + (size_t)(b * N_ + j0 + l15) * C3_ + C_ + h * DH_;
    const v16h bk0 = load_bfrag_contig_h(kr, khalf);
    const v16h bk1 = load_bfrag_contig_h(kr + 32, khalf);
    v8f t = {};
    t = __builtin_amdgcn_wmma_f32_16x16x32_f16(false, aq0, false, bk0, (short)0, t, false, false);
    t = __builtin_amdgcn_wmma_f32_16x16x32_f16(false, aq1, false, bk1, (short)0, t, false, false);
#pragma unroll
    for (int r = 0; r < 8; ++r) ps[r] += __expf(t[r] * SCALE_);
  }

  // reduce over the 16 lanes of each half (rows 0..7 lanes 0-15, 8..15 lanes 16-31)
#pragma unroll
  for (int r = 0; r < 8; ++r) {
    ps[r] += __shfl_xor(ps[r], 1, 32);
    ps[r] += __shfl_xor(ps[r], 2, 32);
    ps[r] += __shfl_xor(ps[r], 4, 32);
    ps[r] += __shfl_xor(ps[r], 8, 32);
  }
  if (l15 == 0) {
    const int rbase = i0 + half * 8;
#pragma unroll
    for (int r = 0; r < 8; ++r)
      lsum[(size_t)bh * N_ + rbase + r] = ps[r];
  }
}

// ---------------------------------------------------------------------------
// Attention output: yin[b, j, h*64+d] = sum_i exp(scale*q_i.k_j) * g_i/l_i * v[i,d]
// plus per-head prune partials prune_part[b,h,j] = sum_i (weights >= 0).
// One wave per (b,h, 16-row j-tile); i in chunks of 32. The probability tile
// goes through wave-private LDS to re-layout C-frag -> A-frag.
// ---------------------------------------------------------------------------
__global__ __launch_bounds__(256)
void attn_out(const _Float16* __restrict__ qkv, const float* __restrict__ g,
              const float* __restrict__ lsum, float* __restrict__ yin,
              float* __restrict__ prune_part) {
  __shared__ _Float16 P[8][16][32];            // per-wave probability tile (8 KB)

  const int wave = (int)((blockIdx.x * blockDim.x + threadIdx.x) >> 5);
  const int wib  = (int)(threadIdx.x >> 5);
  const int lane = (int)(threadIdx.x & 31);
  const int tiles_j = N_ / 16;                 // 36
  const int bh = wave / tiles_j;
  if (bh >= B_ * H_) return;
  const int jtile = wave % tiles_j;
  const int b = bh / H_, h = bh % H_;
  const int j0 = jtile * 16;

  const int l15 = lane & 15, half = lane >> 4;
  const int abase = half * 8, khalf = half * 16;

  // A-fragments: k rows j0..j0+15 (T[j,i] = sum_d k[j,d] q[i,d])
  const _Float16* kr = qkv + (size_t)(b * N_ + j0 + l15) * C3_ + C_ + h * DH_;
  const v16h ak0 = load_afrag_h(kr, abase);
  const v16h ak1 = load_afrag_h(kr + 32, abase);

  v8f acc0 = {}, acc1 = {}, acc2 = {}, acc3 = {};
  float pr[8];
#pragma unroll
  for (int r = 0; r < 8; ++r) pr[r] = 0.0f;

  const float* gp = g    + (size_t)bh * N_;
  const float* lp = lsum + (size_t)bh * N_;

  for (int i0 = 0; i0 < N_; i0 += 32) {
    // ---- score tiles T0 (cols i0..i0+15) and T1 (cols i0+16..i0+31) ----
    v8f t0 = {}, t1 = {};
    {
      const _Float16* qr0 = qkv + (size_t)(b * N_ + i0 + l15) * C3_ + h * DH_;
      const _Float16* qr1 = qr0 + 16 * (size_t)C3_;
      const v16h bq0 = load_bfrag_contig_h(qr0, khalf);
      const v16h bq1 = load_bfrag_contig_h(qr0 + 32, khalf);
      const v16h bq2 = load_bfrag_contig_h(qr1, khalf);
      const v16h bq3 = load_bfrag_contig_h(qr1 + 32, khalf);
      t0 = __builtin_amdgcn_wmma_f32_16x16x32_f16(false, ak0, false, bq0, (short)0, t0, false, false);
      t0 = __builtin_amdgcn_wmma_f32_16x16x32_f16(false, ak1, false, bq1, (short)0, t0, false, false);
      t1 = __builtin_amdgcn_wmma_f32_16x16x32_f16(false, ak0, false, bq2, (short)0, t1, false, false);
      t1 = __builtin_amdgcn_wmma_f32_16x16x32_f16(false, ak1, false, bq3, (short)0, t1, false, false);
    }

    // ---- elementwise exp * g/l ; C-frag col = lane&15, row = r (+8 hi half) ----
    const float ca = gp[i0 + l15]      / lp[i0 + l15];
    const float cb = gp[i0 + 16 + l15] / lp[i0 + 16 + l15];
    const int prow = half * 8;
#pragma unroll
    for (int r = 0; r < 8; ++r) {
      const float e0 = __expf(t0[r] * SCALE_) * ca;
      const float e1 = __expf(t1[r] * SCALE_) * cb;
      pr[r] += e0 + e1;
      P[wib][prow + r][l15]      = (_Float16)e0;
      P[wib][prow + r][l15 + 16] = (_Float16)e1;
    }

    // ---- re-load as A-fragment (rows j, K = local i 0..31); pairs contiguous ----
    const v16h apf = load_afrag_h(&P[wib][l15][0], abase);

    // ---- B-fragments from V (K = local i, N = d tile), accumulate out ----
    const _Float16* vbase = qkv + (size_t)(b * N_ + i0) * C3_ + 2 * C_ + h * DH_;
#pragma unroll
    for (int dt = 0; dt < 4; ++dt) {
      const int d = dt * 16 + l15;
      v16h bv;
#pragma unroll
      for (int v = 0; v < 8; ++v) {
        const int il = khalf + 2 * v;
        bv[2 * v]     = vbase[(size_t)il * C3_ + d];
        bv[2 * v + 1] = vbase[(size_t)(il + 1) * C3_ + d];
      }
      switch (dt) {
        case 0: acc0 = __builtin_amdgcn_wmma_f32_16x16x32_f16(false, apf, false, bv, (short)0, acc0, false, false); break;
        case 1: acc1 = __builtin_amdgcn_wmma_f32_16x16x32_f16(false, apf, false, bv, (short)0, acc1, false, false); break;
        case 2: acc2 = __builtin_amdgcn_wmma_f32_16x16x32_f16(false, apf, false, bv, (short)0, acc2, false, false); break;
        case 3: acc3 = __builtin_amdgcn_wmma_f32_16x16x32_f16(false, apf, false, bv, (short)0, acc3, false, false); break;
      }
    }
  }

  // ---- store out rows into [B, N, C] layout for the proj GEMM ----
  const int jrow = j0 + half * 8;
#pragma unroll
  for (int r = 0; r < 8; ++r) {
    float* yr = yin + (size_t)(b * N_ + jrow + r) * C_ + h * DH_ + l15;
    yr[0]  = acc0[r];
    yr[16] = acc1[r];
    yr[32] = acc2[r];
    yr[48] = acc3[r];
  }

  // ---- prune partials: reduce across 16 lanes, deterministic store per head ----
#pragma unroll
  for (int r = 0; r < 8; ++r) {
    pr[r] += __shfl_xor(pr[r], 1, 32);
    pr[r] += __shfl_xor(pr[r], 2, 32);
    pr[r] += __shfl_xor(pr[r], 4, 32);
    pr[r] += __shfl_xor(pr[r], 8, 32);
  }
  if (l15 == 0) {
#pragma unroll
    for (int r = 0; r < 8; ++r)
      prune_part[(size_t)bh * N_ + jrow + r] = pr[r];
  }
}

// ---------------------------------------------------------------------------
// Deterministic prune reduction: prune[b,j] = sum_h prune_part[b,h,j].
// ---------------------------------------------------------------------------
__global__ __launch_bounds__(256)
void prune_reduce(const float* __restrict__ prune_part, float* __restrict__ prune) {
  const int r = (int)(blockIdx.x * blockDim.x + threadIdx.x);
  if (r >= BN_) return;
  const int b = r / N_, j = r % N_;
  float s = 0.0f;
#pragma unroll
  for (int h = 0; h < H_; ++h)
    s += prune_part[((size_t)(b * H_ + h)) * N_ + j];
  prune[r] = s;
}

// ---------------------------------------------------------------------------
extern "C" void kernel_launch(void* const* d_in, const int* in_sizes, int n_in,
                              void* d_out, int out_size, void* d_ws, size_t ws_size,
                              hipStream_t stream) {
  (void)in_sizes; (void)n_in; (void)out_size; (void)ws_size;
  const float* x      = (const float*)d_in[0];
  const float* m_in   = (const float*)d_in[1];
  const float* W_qkv  = (const float*)d_in[2];
  const float* W_hs   = (const float*)d_in[3];
  const float* W_mc1  = (const float*)d_in[4];
  const float* W_mc2  = (const float*)d_in[5];
  const float* W_proj = (const float*)d_in[6];
  const float* b_proj = (const float*)d_in[7];

  float* out = (float*)d_out;
  float* y_out     = out;                         // B*N*C
  float* mout_out  = out + (size_t)BN_ * C_;      // B*N
  float* prune_out = mout_out + BN_;              // B*N

  // workspace carve (f16 qkv first, then f32 buffers)
  char* wsb = (char*)d_ws;
  _Float16* qkv = (_Float16*)wsb;                          // B*N*3C halfs
  size_t off = (size_t)BN_ * C3_ * sizeof(_Float16);
  float* hs    = (float*)(wsb + off); off += (size_t)BN_ * H_ * sizeof(float);
  float* mc1   = (float*)(wsb + off); off += (size_t)BN_ * (C_ / 2) * sizeof(float);
  float* g     = (float*)(wsb + off); off += (size_t)B_ * H_ * N_ * sizeof(float);
  float* lsum  = (float*)(wsb + off); off += (size_t)B_ * H_ * N_ * sizeof(float);
  float* ppart = (float*)(wsb + off); off += (size_t)B_ * H_ * N_ * sizeof(float);
  float* yin   = (float*)(wsb + off);

  auto gemm_blocks = [](int M, int N) {
    const int tiles = (M / 16) * ((N + 15) / 16);
    return (tiles + 7) / 8;
  };

  // 1) qkv = x @ W_qkv   [9216,768]x[768,2304] -> f16
  gemm16_wmma<false, true><<<gemm_blocks(BN_, C3_), 256, 0, stream>>>(
      x, W_qkv, nullptr, qkv, nullptr, BN_, C3_, C_, C3_ / 16);

  // 2) hs = x @ W_hs     [9216,768]x[768,12]  (column-masked)
  gemm16_wmma<true, false><<<gemm_blocks(BN_, H_), 256, 0, stream>>>(
      x, W_hs, hs, nullptr, nullptr, BN_, H_, C_, 1);

  // 3) mc1 = x @ W_mc1   [9216,768]x[768,384]
  gemm16_wmma<false, false><<<gemm_blocks(BN_, C_ / 2), 256, 0, stream>>>(
      x, W_mc1, mc1, nullptr, nullptr, BN_, C_ / 2, C_, (C_ / 2) / 16);

  // 4) gates
  gate_kernel<<<(BN_ + 255) / 256, 256, 0, stream>>>(hs, mc1, W_mc2, m_in,
                                                     mout_out, g);

  // 5) softmax row sums
  {
    const int waves = B_ * H_ * (N_ / 16);        // 6912
    attn_stats<<<(waves + 7) / 8, 256, 0, stream>>>(qkv, lsum);
  }

  // 6) gated reversed attention + per-head prune partials
  {
    const int waves = B_ * H_ * (N_ / 16);        // 6912
    attn_out<<<(waves + 7) / 8, 256, 0, stream>>>(qkv, g, lsum, yin, ppart);
  }

  // 7) prune_mask[b,j] = sum_h partials (fixed order, deterministic)
  prune_reduce<<<(BN_ + 255) / 256, 256, 0, stream>>>(ppart, prune_out);

  // 8) y = yin @ W_proj + b_proj
  gemm16_wmma<false, false><<<gemm_blocks(BN_, C_), 256, 0, stream>>>(
      yin, W_proj, y_out, nullptr, b_proj, BN_, C_, C_, C_ / 16);
}
